// MultiHeadAttention_59760174957330
// MI455X (gfx1250) — matrix operand order repaired and checked by reading, not compiled, not run on hardware
//
#include <hip/hip_runtime.h>
#include <cstddef>
#include <cstdint>

// ---------------------------------------------------------------------------
// CDNA5 (gfx1250) multi-head attention forward, bf16 WMMA + f32 accumulate.
// Round 2: async global->LDS copy (ASYNCcnt path) for the B tile in the QKV
// GEMM + prefetch of the next A tile.
// ---------------------------------------------------------------------------

typedef __bf16 bf16;
typedef __bf16 v8bf  __attribute__((ext_vector_type(8)));
typedef __bf16 v16bf __attribute__((ext_vector_type(16)));
typedef float  v8f   __attribute__((ext_vector_type(8)));

#define DMODEL   1024
#define SEQ      2048
#define DH       64
#define HEADSZ   131072      /* 2048*64  elements per head chunk   */
#define ATT_HEAD 4194304     /* 2048*2048 attention elems per head */
#define ATT_OFF  4194304     /* 2*2048*1024 normed elems in d_out  */

static __device__ inline v16bf bfcat(v8bf lo, v8bf hi) {
  return __builtin_shufflevector(lo, hi, 0,1,2,3,4,5,6,7,8,9,10,11,12,13,14,15);
}
static __device__ inline v8f wmma_bf16(v16bf a, v16bf b, v8f c) {
  // D = A(16x32 bf16) x B(32x16 bf16) + C(16x16 f32)
  return __builtin_amdgcn_wmma_f32_16x16x32_bf16(false, a, false, b, (short)0, c,
                                                 false, false);
}
static __device__ inline void cvt4_8(v8bf& d, int i, float4 f) {
  d[i] = (bf16)f.x; d[i+1] = (bf16)f.y; d[i+2] = (bf16)f.z; d[i+3] = (bf16)f.w;
}
static __device__ inline void cvt4_16(v16bf& d, int i, float4 f) {
  d[i] = (bf16)f.x; d[i+1] = (bf16)f.y; d[i+2] = (bf16)f.z; d[i+3] = (bf16)f.w;
}

// Per-lane async copy of 16 bytes: global (saddr base + 32-bit byte offset)
// -> LDS (32-bit wave-relative address).  Tracked by ASYNCcnt.
static __device__ inline void async_copy_b128(uint32_t lds_addr,
                                              uint32_t glb_off,
                                              const void* saddr_base) {
  asm volatile("global_load_async_to_lds_b128 %0, %1, %2"
               :
               : "v"(lds_addr), "v"(glb_off), "s"(saddr_base)
               : "memory");
}
static __device__ inline void wait_asynccnt0() {
  asm volatile("s_wait_asynccnt 0x0" ::: "memory");
}
// Generic LDS pointer -> wave-relative LDS byte address (low 32 bits of the
// generic address).
static __device__ inline uint32_t lds_addr32(const void* p) {
  return (uint32_t)(size_t)p;
}

// ---------------------------------------------------------------------------
// K0: W [1024][1024] f32  ->  WT [n][k] bf16 (transposed + converted)
// ---------------------------------------------------------------------------
__global__ __launch_bounds__(256) void transpose_w(const float* __restrict__ W,
                                                   bf16* __restrict__ WT) {
  __shared__ float t[32][33];
  const int n0 = blockIdx.x * 32, k0 = blockIdx.y * 32;
  const int tx = threadIdx.x & 31, ty = threadIdx.x >> 5;
  for (int p = 0; p < 4; ++p)
    t[ty + 8*p][tx] = W[(size_t)(k0 + ty + 8*p) * DMODEL + n0 + tx];
  __syncthreads();
  for (int p = 0; p < 4; ++p)
    WT[(size_t)(n0 + ty + 8*p) * DMODEL + k0 + tx] = (bf16)t[tx][ty + 8*p];
}

// ---------------------------------------------------------------------------
// K1: q/k/v projections.  Y = X @ W + b.  128x128 tile per workgroup,
// 8 waves of 64x32, K staged through LDS in 32-wide steps (256 WMMAs/wave).
// B tile (bf16 W^T) is staged with GLOBAL_LOAD_ASYNC_TO_LDS_B128 (ASYNCcnt);
// A tile needs f32->bf16 conversion so it goes through VGPRs, with a
// prefetch of the next k-tile overlapping the current WMMAs.
// mode 0 -> q (natural bf16), 1 -> k (natural bf16), 2 -> v (per-head
// transposed bf16 [bh][64][2048], feeding the context GEMM B-operand).
// ---------------------------------------------------------------------------
__global__ __launch_bounds__(256) void qkv_gemm(
    const float* __restrict__ Xq, const float* __restrict__ Xk,
    const float* __restrict__ Xv,
    const bf16* __restrict__ WqT, const bf16* __restrict__ WkT,
    const bf16* __restrict__ WvT,
    const float* __restrict__ bq, const float* __restrict__ bk,
    const float* __restrict__ bv,
    bf16* __restrict__ qo, bf16* __restrict__ ko, bf16* __restrict__ vTo) {
  const int mode = blockIdx.z;
  const float* X    = (mode == 0) ? Xq  : (mode == 1) ? Xk  : Xv;
  const bf16*  WT   = (mode == 0) ? WqT : (mode == 1) ? WkT : WvT;
  const float* bias = (mode == 0) ? bq  : (mode == 1) ? bk  : bv;

  const int m0 = blockIdx.y * 128;
  const int n0 = blockIdx.x * 128;
  const int tid = threadIdx.x;
  const int wave = tid >> 5, lane = tid & 31;
  const int l16 = lane & 15, lh = lane >> 4;
  const int wm = wave >> 2, wn = wave & 3;

  __shared__ bf16 As[128 * 40];   // [row][32+pad]
  __shared__ bf16 Bs[128 * 40];   // WT rows (n-major), [n][32+pad]

  v8f acc[4][2] = {};

  const int lrow = tid >> 1;      // 2 threads per row, 16 elems each
  const int lhalf = tid & 1;

  const float*   arow_g  = X + (size_t)(m0 + lrow) * DMODEL + lhalf * 16;
  const uint32_t bs_lds  = lds_addr32(&Bs[lrow * 40 + lhalf * 16]);
  const uint32_t bg_base = (uint32_t)(((n0 + lrow) * DMODEL + lhalf * 16) *
                                      (int)sizeof(bf16));

  for (int kt = 0; kt < 32; ++kt) {
    const int k0 = kt * 32;
    __syncthreads();
    // B: bf16 W^T tile, direct async copy into LDS (no VGPR round-trip).
    async_copy_b128(bs_lds, bg_base + (uint32_t)(k0 * sizeof(bf16)), WT);
    {
      // A: f32 global -> bf16 LDS (VALU conversion path)
      const float4* s = (const float4*)(arow_g + k0);
      float4 f0 = s[0], f1 = s[1], f2 = s[2], f3 = s[3];
      v8bf lo, hi;
      cvt4_8(lo, 0, f0); cvt4_8(lo, 4, f1);
      cvt4_8(hi, 0, f2); cvt4_8(hi, 4, f3);
      *(v8bf*)&As[lrow * 40 + lhalf * 16]     = lo;
      *(v8bf*)&As[lrow * 40 + lhalf * 16 + 8] = hi;
    }
    if (kt < 31)  // overlap next A tile's HBM latency with this step's WMMAs
      __builtin_prefetch(arow_g + k0 + 32, 0, 1);
    wait_asynccnt0();
    __syncthreads();

    v16bf a[4], b[2];
    for (int mt = 0; mt < 4; ++mt) {
      const int base = (wm * 64 + mt * 16 + l16) * 40 + lh * 8;
      a[mt] = bfcat(*(const v8bf*)&As[base], *(const v8bf*)&As[base + 16]);
    }
    for (int nt = 0; nt < 2; ++nt) {
      const int base = (wn * 32 + nt * 16 + l16) * 40 + lh * 8;
      b[nt] = bfcat(*(const v8bf*)&Bs[base], *(const v8bf*)&Bs[base + 16]);
    }
    for (int mt = 0; mt < 4; ++mt)
      for (int nt = 0; nt < 2; ++nt)
        acc[mt][nt] = wmma_bf16(a[mt], b[nt], acc[mt][nt]);
  }

  for (int nt = 0; nt < 2; ++nt) {
    const int n = n0 + wn * 32 + nt * 16 + l16;
    const float bn = bias[n];
    for (int mt = 0; mt < 4; ++mt) {
      const int mb = m0 + wm * 64 + mt * 16 + 8 * lh;
      for (int r = 0; r < 8; ++r) {
        const int m = mb + r;
        const float v = acc[mt][nt][r] + bn;
        if (mode < 2) {
          bf16* o = (mode == 0) ? qo : ko;
          o[(size_t)m * DMODEL + n] = (bf16)v;
        } else {
          const int flat = m * DMODEL + n;            // == bh*HEADSZ + s*64 + dd
          const int bh = flat >> 17;
          const int rem = flat & (HEADSZ - 1);
          const int s = rem >> 6, dd = rem & 63;
          vTo[(size_t)bh * HEADSZ + dd * SEQ + s] = (bf16)v;
        }
      }
    }
  }
}

// ---------------------------------------------------------------------------
// K2: scores = 0.5 * q @ k^T, fused row softmax, write probabilities f32.
// Workgroup = (head, 16 q-rows); wave w owns key columns [w*256, w*256+256).
// ---------------------------------------------------------------------------
__global__ __launch_bounds__(256) void attn_scores(const bf16* __restrict__ qb,
                                                   const bf16* __restrict__ kb,
                                                   float* __restrict__ att) {
  const int bh = blockIdx.y, qblk = blockIdx.x;
  const int tid = threadIdx.x;
  const int wave = tid >> 5, lane = tid & 31;
  const int l16 = lane & 15, lh = lane >> 4;

  __shared__ bf16 Qs[16 * 72];
  __shared__ float redmax[8 * 16];
  __shared__ float redsum[8 * 16];

  const bf16* qh = qb + (size_t)bh * HEADSZ + (size_t)qblk * 16 * DH;
  const bf16* kh = kb + (size_t)bh * HEADSZ;

  if (tid < 64) {
    const int row = tid >> 2, c = (tid & 3) * 16;
    const v8bf* s = (const v8bf*)(qh + row * DH + c);
    *(v8bf*)&Qs[row * 72 + c]     = s[0];
    *(v8bf*)&Qs[row * 72 + c + 8] = s[1];
  }
  __syncthreads();

  v16bf a[2];
  for (int kk = 0; kk < 2; ++kk) {
    const int base = l16 * 72 + kk * 32 + lh * 8;
    a[kk] = bfcat(*(const v8bf*)&Qs[base], *(const v8bf*)&Qs[base + 16]);
  }

  v8f sc[16];
  const int seqbase = wave * 256;
  for (int j = 0; j < 16; ++j) {
    v8f z = {0.f, 0.f, 0.f, 0.f, 0.f, 0.f, 0.f, 0.f};
    const int seq = seqbase + j * 16 + l16;
    for (int kk = 0; kk < 2; ++kk) {
      const v8bf* p = (const v8bf*)(kh + (size_t)seq * DH + kk * 32 + lh * 8);
      z = wmma_bf16(a[kk], bfcat(p[0], p[2]), z);
    }
    for (int r = 0; r < 8; ++r) z[r] *= 0.5f;   // (d//H)^-0.5 = 0.5
    sc[j] = z;
  }

  // row max (rows 0-7 live in lanes 0-15, rows 8-15 in lanes 16-31)
  float mx[8];
  for (int r = 0; r < 8; ++r) mx[r] = sc[0][r];
  for (int j = 1; j < 16; ++j)
    for (int r = 0; r < 8; ++r) mx[r] = fmaxf(mx[r], sc[j][r]);
  for (int off = 8; off >= 1; off >>= 1)
    for (int r = 0; r < 8; ++r) mx[r] = fmaxf(mx[r], __shfl_xor(mx[r], off, 32));
  if (l16 == 0)
    for (int r = 0; r < 8; ++r) redmax[wave * 16 + lh * 8 + r] = mx[r];
  __syncthreads();
  for (int r = 0; r < 8; ++r) {
    float m = -1e30f;
    const int row = lh * 8 + r;
    for (int w = 0; w < 8; ++w) m = fmaxf(m, redmax[w * 16 + row]);
    mx[r] = m;
  }

  float sm[8] = {0, 0, 0, 0, 0, 0, 0, 0};
  for (int j = 0; j < 16; ++j)
    for (int r = 0; r < 8; ++r) {
      const float e = __expf(sc[j][r] - mx[r]);
      sc[j][r] = e;
      sm[r] += e;
    }
  for (int off = 8; off >= 1; off >>= 1)
    for (int r = 0; r < 8; ++r) sm[r] += __shfl_xor(sm[r], off, 32);
  if (l16 == 0)
    for (int r = 0; r < 8; ++r) redsum[wave * 16 + lh * 8 + r] = sm[r];
  __syncthreads();
  float inv[8];
  for (int r = 0; r < 8; ++r) {
    float s = 0.f;
    const int row = lh * 8 + r;
    for (int w = 0; w < 8; ++w) s += redsum[w * 16 + row];
    inv[r] = 1.0f / s;
  }

  float* ah = att + (size_t)bh * ATT_HEAD;
  for (int j = 0; j < 16; ++j) {
    const int col = seqbase + j * 16 + l16;
    for (int r = 0; r < 8; ++r) {
      const int qrow = qblk * 16 + lh * 8 + r;
      ah[(size_t)qrow * SEQ + col] = sc[j][r] * inv[r];
    }
  }
}

// ---------------------------------------------------------------------------
// K3: context = P @ V.  A = probabilities (f32 global -> bf16 in-register),
// B = vT[bh][64][2048].  Wave computes a 16x64 strip, K loop of 64 steps.
// ---------------------------------------------------------------------------
__global__ __launch_bounds__(256) void attn_context(const float* __restrict__ att,
                                                    const bf16* __restrict__ vT,
                                                    bf16* __restrict__ ctx) {
  const int bh = blockIdx.y;
  const int tid = threadIdx.x;
  const int wave = tid >> 5, lane = tid & 31;
  const int l16 = lane & 15, lh = lane >> 4;
  const int s0 = blockIdx.x * 128 + wave * 16;

  const float* prow = att + (size_t)bh * ATT_HEAD + (size_t)(s0 + l16) * SEQ;
  const bf16* vh = vT + (size_t)bh * HEADSZ;

  v8f acc[4] = {};
  for (int kt = 0; kt < 64; ++kt) {
    const int kb0 = kt * 32 + lh * 8;
    const float4* p0 = (const float4*)(prow + kb0);
    const float4* p1 = (const float4*)(prow + kb0 + 16);
    v16bf a;
    cvt4_16(a, 0, p0[0]); cvt4_16(a, 4, p0[1]);
    cvt4_16(a, 8, p1[0]); cvt4_16(a, 12, p1[1]);
    for (int nt = 0; nt < 4; ++nt) {
      const int n = nt * 16 + l16;
      const v8bf* pb = (const v8bf*)(vh + (size_t)n * SEQ + kt * 32 + lh * 8);
      acc[nt] = wmma_bf16(a, bfcat(pb[0], pb[2]), acc[nt]);
    }
  }
  for (int nt = 0; nt < 4; ++nt) {
    const int dd = nt * 16 + l16;
    for (int r = 0; r < 8; ++r) {
      const int s = s0 + r + 8 * lh;
      ctx[(size_t)bh * HEADSZ + (size_t)s * DH + dd] = (bf16)acc[nt][r];
    }
  }
}

// ---------------------------------------------------------------------------
// K4: out = ctx @ Wf + bf; x = residual + out; layernorm -> d_out.
// Workgroup owns 16 rows x 1024 cols; wave w owns cols [w*128, w*128+128).
// ---------------------------------------------------------------------------
__global__ __launch_bounds__(256) void out_proj_ln(
    const bf16* __restrict__ ctx, const bf16* __restrict__ WfT,
    const float* __restrict__ bfv, const float* __restrict__ residual,
    const float* __restrict__ g, const float* __restrict__ bb,
    float* __restrict__ outp) {
  const int m0 = blockIdx.x * 16;
  const int tid = threadIdx.x;
  const int wave = tid >> 5, lane = tid & 31;
  const int l16 = lane & 15, lh = lane >> 4;
  const int n0w = wave * 128;

  __shared__ float reds[8 * 16];
  __shared__ float redq[8 * 16];

  const bf16* arow = ctx + (size_t)(m0 + l16) * DMODEL;

  v8f acc[8] = {};
  for (int kt = 0; kt < 32; ++kt) {
    const v8bf* pa0 = (const v8bf*)(arow + kt * 32 + lh * 8);
    const v8bf* pa1 = (const v8bf*)(arow + kt * 32 + lh * 8 + 16);
    const v16bf a = bfcat(pa0[0], pa1[0]);
    for (int nt = 0; nt < 8; ++nt) {
      const int n = n0w + nt * 16 + l16;
      const v8bf* pb = (const v8bf*)(WfT + (size_t)n * DMODEL + kt * 32 + lh * 8);
      acc[nt] = wmma_bf16(a, bfcat(pb[0], pb[2]), acc[nt]);
    }
  }

  float s[8]  = {0, 0, 0, 0, 0, 0, 0, 0};
  float q2[8] = {0, 0, 0, 0, 0, 0, 0, 0};
  for (int nt = 0; nt < 8; ++nt) {
    const int n = n0w + nt * 16 + l16;
    const float bn = bfv[n];
    for (int r = 0; r < 8; ++r) {
      const int m = m0 + r + 8 * lh;
      const float x = acc[nt][r] + bn + residual[(size_t)m * DMODEL + n];
      acc[nt][r] = x;
      s[r] += x;
      q2[r] += x * x;
    }
  }
  for (int off = 8; off >= 1; off >>= 1)
    for (int r = 0; r < 8; ++r) {
      s[r]  += __shfl_xor(s[r], off, 32);
      q2[r] += __shfl_xor(q2[r], off, 32);
    }
  if (l16 == 0)
    for (int r = 0; r < 8; ++r) {
      reds[wave * 16 + lh * 8 + r] = s[r];
      redq[wave * 16 + lh * 8 + r] = q2[r];
    }
  __syncthreads();
  float mean[8], rstd[8];
  for (int r = 0; r < 8; ++r) {
    const int row = lh * 8 + r;
    float ts = 0.f, tq = 0.f;
    for (int w = 0; w < 8; ++w) { ts += reds[w * 16 + row]; tq += redq[w * 16 + row]; }
    const float mu = ts * (1.0f / DMODEL);
    const float var = tq * (1.0f / DMODEL) - mu * mu;
    mean[r] = mu;
    rstd[r] = rsqrtf(var + 1e-5f);
  }
  for (int nt = 0; nt < 8; ++nt) {
    const int n = n0w + nt * 16 + l16;
    const float gg = g[n], b2 = bb[n];
    for (int r = 0; r < 8; ++r) {
      const int m = m0 + r + 8 * lh;
      outp[(size_t)m * DMODEL + n] = (acc[nt][r] - mean[r]) * rstd[r] * gg + b2;
    }
  }
}

// ---------------------------------------------------------------------------
// Launch.  Workspace (40 MB): 4x bf16 W^T (8 MB) | q bf16 | k bf16 | v^T bf16
// | ctx bf16 (8 MB each).
// ---------------------------------------------------------------------------
extern "C" void kernel_launch(void* const* d_in, const int* in_sizes, int n_in,
                              void* d_out, int out_size, void* d_ws, size_t ws_size,
                              hipStream_t stream) {
  (void)in_sizes; (void)n_in; (void)out_size; (void)ws_size;

  const float* key   = (const float*)d_in[0];
  const float* value = (const float*)d_in[1];
  const float* query = (const float*)d_in[2];
  const float* Wk    = (const float*)d_in[3];
  const float* bk    = (const float*)d_in[4];
  const float* Wv    = (const float*)d_in[5];
  const float* bv    = (const float*)d_in[6];
  const float* Wq    = (const float*)d_in[7];
  const float* bq    = (const float*)d_in[8];
  const float* Wf    = (const float*)d_in[9];
  const float* bfv   = (const float*)d_in[10];
  const float* ln_g  = (const float*)d_in[11];
  const float* ln_b  = (const float*)d_in[12];

  char* ws = (char*)d_ws;
  bf16* WKT = (bf16*)(ws + 0ul * (1ul << 21));
  bf16* WVT = (bf16*)(ws + 1ul * (1ul << 21));
  bf16* WQT = (bf16*)(ws + 2ul * (1ul << 21));
  bf16* WFT = (bf16*)(ws + 3ul * (1ul << 21));
  bf16* QB  = (bf16*)(ws + (8ul  << 20));
  bf16* KB  = (bf16*)(ws + (16ul << 20));
  bf16* VT  = (bf16*)(ws + (24ul << 20));
  bf16* CTX = (bf16*)(ws + (32ul << 20));

  float* normed = (float*)d_out;
  float* att    = (float*)d_out + ATT_OFF;

  const dim3 tb(256);
  transpose_w<<<dim3(32, 32), tb, 0, stream>>>(Wk, WKT);
  transpose_w<<<dim3(32, 32), tb, 0, stream>>>(Wv, WVT);
  transpose_w<<<dim3(32, 32), tb, 0, stream>>>(Wq, WQT);
  transpose_w<<<dim3(32, 32), tb, 0, stream>>>(Wf, WFT);

  qkv_gemm<<<dim3(8, 32, 3), tb, 0, stream>>>(query, key, value,
                                              WQT, WKT, WVT,
                                              bq, bk, bv,
                                              QB, KB, VT);

  attn_scores<<<dim3(128, 32), tb, 0, stream>>>(QB, KB, att);
  attn_context<<<dim3(16, 32), tb, 0, stream>>>(att, VT, CTX);
  out_proj_ln<<<dim3(256), tb, 0, stream>>>(CTX, WFT, bfv, query, ln_g, ln_b, normed);
}
